// PunishGraphAttentionLayer_28174985462251
// MI455X (gfx1250) — matrix operand order repaired and checked by reading, not compiled, not run on hardware
//
#include <hip/hip_runtime.h>
#include <math.h>

// Problem sizes (fixed by the reference)
#define GAT_BS   32
#define GAT_N    1024
#define GAT_INF  512
#define GAT_OUTF 128
#define GAT_ALPHA 0.2f
#define GAT_NEGBIG -9e15f

typedef __attribute__((ext_vector_type(16))) __bf16 v16bf;
typedef __attribute__((ext_vector_type(8)))  float  v8f;

union BF16x16 { v16bf v; unsigned u[8]; };
union F32x8   { v8f  v; float    f[8]; };

__device__ __forceinline__ unsigned bf16rne(float x) {
  unsigned u = __float_as_uint(x);
  return (u + 0x7fffu + ((u >> 16) & 1u)) >> 16;   // round-to-nearest-even
}
__device__ __forceinline__ unsigned packbf2(float lo, float hi) {
  return (bf16rne(hi) << 16) | (bf16rne(lo) & 0xffffu);
}

// ---------------------------------------------------------------------------
// Kernel 1: Wh = h @ W + b   (bf16 WMMA, f32 accumulate)
// Fused epilogue: f1 = Wh.a1, f2 = Wh.a2 (per-row), and Whs = Wh*scores
// written as packed bf16 pairs in the B-matrix layout needed by kernel 2:
//   whs[(b*128 + col)*512 + j/2] = { bf16(Whs[j+1][col]), bf16(Whs[j][col]) }
// ---------------------------------------------------------------------------
__global__ __launch_bounds__(256)
void gat_wh_gemm(const float* __restrict__ hmat, const float* __restrict__ Ww,
                 const float* __restrict__ Wb,   const float* __restrict__ scores,
                 const float* __restrict__ a1,   const float* __restrict__ a2,
                 unsigned* __restrict__ whs, float* __restrict__ f1,
                 float* __restrict__ f2)
{
  // Packed bf16-pair tiles: lA[row][kk] = {h[k0+2kk+1], h[k0+2kk]}
  __shared__ unsigned lA[128][17];   // +1 pad: 17 odd -> conflict-free col reads
  __shared__ unsigned lB[16][128];   // lB[kk][col] = {W[k0+2kk+1][col], W[k0+2kk][col]}

  const int tid  = threadIdx.x;
  const int w    = tid >> 5;        // wave id (8 waves)
  const int l    = tid & 31;
  const int lane = l & 15;
  const int hh   = l >> 4;          // lane half
  const size_t rowbase = (size_t)blockIdx.x * 128;   // 256 blocks cover 32768 rows

  F32x8 acc[8];
#pragma unroll
  for (int n = 0; n < 8; ++n)
#pragma unroll
    for (int k = 0; k < 8; ++k) acc[n].f[k] = 0.f;

  const int rlA = tid >> 1, khA = (tid & 1) * 8;     // A staging assignment
  const int kkB = tid >> 4, c0B = (tid & 15) * 8;    // B staging assignment

  for (int kt = 0; kt < GAT_INF / 32; ++kt) {
    const int k0 = kt * 32;
    { // stage A: 128 rows x 32 cols f32 -> bf16 pairs (row-contiguous loads)
      const float4* s4 = (const float4*)(hmat + (rowbase + rlA) * GAT_INF + k0 + khA * 2);
      float4 x0 = s4[0], x1 = s4[1], x2 = s4[2], x3 = s4[3];
      lA[rlA][khA + 0] = packbf2(x0.x, x0.y); lA[rlA][khA + 1] = packbf2(x0.z, x0.w);
      lA[rlA][khA + 2] = packbf2(x1.x, x1.y); lA[rlA][khA + 3] = packbf2(x1.z, x1.w);
      lA[rlA][khA + 4] = packbf2(x2.x, x2.y); lA[rlA][khA + 5] = packbf2(x2.z, x2.w);
      lA[rlA][khA + 6] = packbf2(x3.x, x3.y); lA[rlA][khA + 7] = packbf2(x3.z, x3.w);
    }
    { // stage B: 32 rows (K) x 128 cols of W, packed pairwise along K
      const float*  r0 = Ww + (size_t)(k0 + 2 * kkB) * GAT_OUTF + c0B;
      const float4* p0 = (const float4*)r0;
      const float4* p1 = (const float4*)(r0 + GAT_OUTF);
      float4 a0 = p0[0], a1q = p0[1], b0 = p1[0], b1q = p1[1];
      lB[kkB][c0B + 0] = packbf2(a0.x,  b0.x);  lB[kkB][c0B + 1] = packbf2(a0.y,  b0.y);
      lB[kkB][c0B + 2] = packbf2(a0.z,  b0.z);  lB[kkB][c0B + 3] = packbf2(a0.w,  b0.w);
      lB[kkB][c0B + 4] = packbf2(a1q.x, b1q.x); lB[kkB][c0B + 5] = packbf2(a1q.y, b1q.y);
      lB[kkB][c0B + 6] = packbf2(a1q.z, b1q.z); lB[kkB][c0B + 7] = packbf2(a1q.w, b1q.w);
    }
    __syncthreads();

    // A fragment: lane = row (both halves), VGPR r holds K = kbase(r,hh)+{0,1}
    BF16x16 af;
    const int arow = 16 * w + lane;
#pragma unroll
    for (int r = 0; r < 8; ++r) {
      int kk = ((r < 4) ? r : r + 4) + 4 * hh;   // kbase/2
      af.u[r] = lA[arow][kk];
    }
#pragma unroll
    for (int n = 0; n < 8; ++n) {
      BF16x16 bfg;  // B fragment: col = lane, K = kbase(r,hh) + {0,1}
#pragma unroll
      for (int r = 0; r < 8; ++r) bfg.u[r] = lB[((r < 4) ? r : r + 4) + 4 * hh][16 * n + lane];
      acc[n].v = __builtin_amdgcn_wmma_f32_16x16x32_bf16(
          false, af.v, false, bfg.v, (short)0, acc[n].v, false, false);
    }
    __syncthreads();
  }

  // Epilogue: C layout => element r of acc is row (r + 8*hh), col = 16n + lane
  const int i0    = (int)rowbase + 16 * w;
  const int b     = i0 >> 10;           // batch (blocks never straddle batches)
  const int irow0 = i0 & (GAT_N - 1);
  float a1v[8], a2v[8], bv[8];
#pragma unroll
  for (int n = 0; n < 8; ++n) {
    int c = 16 * n + lane;
    a1v[n] = a1[c]; a2v[n] = a2[c]; bv[n] = Wb[c];
  }
#pragma unroll
  for (int m = 0; m < 4; ++m) {
    float wsv[2][8];
#pragma unroll
    for (int pp = 0; pp < 2; ++pp) {
      const int r    = 2 * m + pp;
      const int grow = i0 + r + 8 * hh;
      const float sc = scores[grow];
      float s1 = 0.f, s2 = 0.f;
#pragma unroll
      for (int n = 0; n < 8; ++n) {
        float wv = acc[n].f[r] + bv[n];
        s1 += wv * a1v[n];
        s2 += wv * a2v[n];
        wsv[pp][n] = wv * sc;
      }
      // reduce across the 16 lanes of this half (rows differ between halves)
      s1 += __shfl_xor(s1, 1, 32); s2 += __shfl_xor(s2, 1, 32);
      s1 += __shfl_xor(s1, 2, 32); s2 += __shfl_xor(s2, 2, 32);
      s1 += __shfl_xor(s1, 4, 32); s2 += __shfl_xor(s2, 4, 32);
      s1 += __shfl_xor(s1, 8, 32); s2 += __shfl_xor(s2, 8, 32);
      if (lane == 0) { f1[grow] = s1; f2[grow] = s2; }
    }
    // rows 2m and 2m+1 (same lane) -> one packed bf16 pair per column
    const int jp = (irow0 + 8 * hh + 2 * m) >> 1;
#pragma unroll
    for (int n = 0; n < 8; ++n) {
      int c = 16 * n + lane;
      whs[((size_t)b * GAT_OUTF + c) * (GAT_N / 2) + jp] = packbf2(wsv[0][n], wsv[1][n]);
    }
  }
}

// ---------------------------------------------------------------------------
// Kernel 2: flash-attention style fused softmax + (att @ Whs) + ELU.
// One wave owns a 16-row tile; 32 column steps of 32; never materializes att.
// ---------------------------------------------------------------------------
__global__ __launch_bounds__(128)
void gat_flash_att(const int* __restrict__ adj, const unsigned* __restrict__ whs,
                   const float* __restrict__ f1, const float* __restrict__ f2,
                   const float* __restrict__ abp, float* __restrict__ out)
{
  const int tid  = threadIdx.x;
  const int w    = tid >> 5;
  const int l    = tid & 31;
  const int lane = l & 15;
  const int hh   = l >> 4;
  const int tile = blockIdx.x * 4 + w;         // 2048 tiles of 16 rows
  const int i0   = tile * 16;
  const int b    = i0 >> 10;
  const int irow0 = i0 & (GAT_N - 1);
  const float ab  = abp[0];

  const float f1v = f1[i0 + lane];             // P-row for this lane = lane
  const float* f2b = f2 + (size_t)b * GAT_N;
  const int* adjrow = adj + ((size_t)b * GAT_N + irow0 + lane) * GAT_N;
  const unsigned* whsb = whs + (size_t)b * GAT_OUTF * (GAT_N / 2);

  F32x8 acc[8];
#pragma unroll
  for (int n = 0; n < 8; ++n)
#pragma unroll
    for (int k = 0; k < 8; ++k) acc[n].f[k] = 0.f;

  float mrun = -INFINITY, lrun = 0.f;

  for (int jt = 0; jt < GAT_N / 32; ++jt) {
    const int j0 = jt * 32;
    if (jt + 1 < GAT_N / 32)
      __builtin_prefetch(adjrow + j0 + 32, 0, 0);   // global_prefetch next adj tile

    // A-fragment element order: e[2r+p] <-> K = kbase(r,hh)+p
    //   elements 0..7  cover K = 8*hh .. 8*hh+7
    //   elements 8..15 cover K = 16+8*hh .. 16+8*hh+7
    const int4*   a4 = (const int4*)(adjrow + j0 + 8 * hh);
    const float4* g4 = (const float4*)(f2b + j0 + 8 * hh);
    int4   q0 = a4[0], q1 = a4[1], q2 = a4[4], q3 = a4[5];
    float4 g0 = g4[0], g1 = g4[1], g2 = g4[4], g3 = g4[5];
    int   av[16] = {q0.x,q0.y,q0.z,q0.w, q1.x,q1.y,q1.z,q1.w,
                    q2.x,q2.y,q2.z,q2.w, q3.x,q3.y,q3.z,q3.w};
    float ev[16] = {g0.x,g0.y,g0.z,g0.w, g1.x,g1.y,g1.z,g1.w,
                    g2.x,g2.y,g2.z,g2.w, g3.x,g3.y,g3.z,g3.w};

    float tmax = -INFINITY;
#pragma unroll
    for (int i = 0; i < 16; ++i) {
      float e = f1v + ev[i] + ab;
      e = (e >= 0.f) ? e : GAT_ALPHA * e;       // leaky_relu
      e = (av[i] > 0) ? e : GAT_NEGBIG;         // adjacency mask
      ev[i] = e;
      tmax = fmaxf(tmax, e);
    }
    // row stats combine across the two halves (same row, different K ranges)
    tmax = fmaxf(tmax, __shfl_xor(tmax, 16, 32));
    const float mnew  = fmaxf(mrun, tmax);
    const float scale = __expf(mrun - mnew);
    float ps = 0.f;
#pragma unroll
    for (int i = 0; i < 16; ++i) { float p = __expf(ev[i] - mnew); ev[i] = p; ps += p; }
    ps += __shfl_xor(ps, 16, 32);
    lrun = lrun * scale + ps;
    mrun = mnew;

    // rescale accumulators: C-row of element r is (r + 8*hh); its scale lives
    // in lane (r + 8*hh) (P-row == lane there)
    float scr[8];
#pragma unroll
    for (int r = 0; r < 8; ++r) scr[r] = __shfl(scale, r + 8 * hh, 32);
#pragma unroll
    for (int n = 0; n < 8; ++n)
#pragma unroll
      for (int r = 0; r < 8; ++r) acc[n].f[r] *= scr[r];

    // P tile -> bf16 A fragment
    BF16x16 pf;
#pragma unroll
    for (int r = 0; r < 8; ++r) pf.u[r] = packbf2(ev[2 * r], ev[2 * r + 1]);

    // B fragments straight from the pre-packed Whs layout: one dword per VGPR
    const int kkb = (j0 >> 1) + 4 * hh;
#pragma unroll
    for (int n = 0; n < 8; ++n) {
      BF16x16 bu;
      const unsigned* bp = whsb + (size_t)(16 * n + lane) * (GAT_N / 2) + kkb;
#pragma unroll
      for (int r = 0; r < 8; ++r) bu.u[r] = bp[(r < 4) ? r : r + 4];
      acc[n].v = __builtin_amdgcn_wmma_f32_16x16x32_bf16(
          false, pf.v, false, bu.v, (short)0, acc[n].v, false, false);
    }
  }

  // normalize, ELU, store (coalesced over the 16 lanes of each half)
#pragma unroll
  for (int r = 0; r < 8; ++r) {
    const float lr  = __shfl(lrun, r + 8 * hh, 32);
    const float inv = 1.f / lr;
    const int grow  = i0 + r + 8 * hh;
#pragma unroll
    for (int n = 0; n < 8; ++n) {
      float x = acc[n].f[r] * inv;
      x = (x > 0.f) ? x : (__expf(x) - 1.f);    // elu
      out[(size_t)grow * GAT_OUTF + 16 * n + lane] = x;
    }
  }
}

// ---------------------------------------------------------------------------
extern "C" void kernel_launch(void* const* d_in, const int* in_sizes, int n_in,
                              void* d_out, int out_size, void* d_ws, size_t ws_size,
                              hipStream_t stream)
{
  const float* hmat   = (const float*)d_in[0];
  const int*   adj    = (const int*)  d_in[1];
  const float* scores = (const float*)d_in[2];
  const float* Ww     = (const float*)d_in[3];
  const float* Wb     = (const float*)d_in[4];
  const float* a1     = (const float*)d_in[5];
  const float* a2     = (const float*)d_in[6];
  const float* ab     = (const float*)d_in[7];
  float* out = (float*)d_out;

  // workspace layout: [whs packed bf16 pairs: 8MB][f1: 128KB][f2: 128KB]
  unsigned* whs = (unsigned*)d_ws;
  float* f1 = (float*)((char*)d_ws + (size_t)GAT_BS * GAT_OUTF * (GAT_N / 2) * sizeof(unsigned));
  float* f2 = f1 + (size_t)GAT_BS * GAT_N;

  gat_wh_gemm<<<dim3(256), dim3(256), 0, stream>>>(hmat, Ww, Wb, scores, a1, a2,
                                                   whs, f1, f2);
  gat_flash_att<<<dim3(512), dim3(128), 0, stream>>>(adj, whs, f1, f2, ab, out);
}
